// Bert_BiLSTM_CRF_16329465660024
// MI455X (gfx1250) — matrix operand order repaired and checked by reading, not compiled, not run on hardware
//
#include <hip/hip_runtime.h>

typedef __attribute__((ext_vector_type(16))) _Float16 v16h;
typedef __attribute__((ext_vector_type(8)))  _Float16 v8h;
typedef __attribute__((ext_vector_type(8)))  float    v8f;
typedef __attribute__((ext_vector_type(4)))  float    v4f;

// Problem constants
constexpr int cB  = 64;     // batch
constexpr int cT  = 512;    // time
constexpr int cD  = 768;    // embed dim
constexpr int cH  = 128;    // hidden
constexpr int cG  = 512;    // 4*H gates
constexpr int cK2 = 256;    // 2*H
constexpr int cNK = 11;     // tags
constexpr int cM  = cT * cB; // 32768 GEMM rows

// ---------------------------------------------------------------------------
// WMMA fragment helpers (V_WMMA_F32_16X16X32_F16 layouts, ISA 7.12.2)
// ---------------------------------------------------------------------------
__device__ __forceinline__ v16h frag_a_h(const _Float16* rp, int k, int half) {
  // A 16x32 f16: j<8 -> K=8*half+j ; j>=8 -> K=16+8*half+(j-8)
  const _Float16* p0 = rp + k + 8 * half;
  v8h lo = *(const v8h*)p0;
  v8h hi = *(const v8h*)(p0 + 16);
  v16h a;
#pragma unroll
  for (int j = 0; j < 8; ++j) { a[j] = lo[j]; a[8 + j] = hi[j]; }
  return a;
}

__device__ __forceinline__ v16h frag_a_f(const float* rp, int k, int half) {
  // same layout, converting f32 -> f16 on the fly
  const float* p0 = rp + k + 8 * half;
  v4f x0 = *(const v4f*)(p0);
  v4f x1 = *(const v4f*)(p0 + 4);
  v4f x2 = *(const v4f*)(p0 + 16);
  v4f x3 = *(const v4f*)(p0 + 20);
  v16h a;
#pragma unroll
  for (int j = 0; j < 4; ++j) {
    a[j]      = (_Float16)x0[j];
    a[4 + j]  = (_Float16)x1[j];
    a[8 + j]  = (_Float16)x2[j];
    a[12 + j] = (_Float16)x3[j];
  }
  return a;
}

__device__ __forceinline__ v16h frag_b_h(const _Float16* colp, int k, int half) {
  // B 32x16 f16: j<8 -> K=8*half+j ; j>=8 -> K=16+8*half+(j-8)
  const _Float16* p0 = colp + k + 8 * half;
  v8h lo = *(const v8h*)p0;
  v8h hi = *(const v8h*)(p0 + 16);
  v16h b;
#pragma unroll
  for (int j = 0; j < 8; ++j) { b[j] = lo[j]; b[8 + j] = hi[j]; }
  return b;
}

__device__ __forceinline__ v8f wmma_f16(v16h a, v16h b, v8f c) {
  return __builtin_amdgcn_wmma_f32_16x16x32_f16(false, a, false, b, (short)0, c,
                                                false, false);
}

__device__ __forceinline__ float sigf(float x) {
  return 1.0f / (1.0f + __expf(-x));
}

// ---------------------------------------------------------------------------
// f32 -> f16 weight conversion
// ---------------------------------------------------------------------------
__global__ void k_f32_to_f16(const float* __restrict__ src,
                             _Float16* __restrict__ dst, int n) {
  int i = blockIdx.x * blockDim.x + threadIdx.x;
  if (i < n) dst[i] = (_Float16)src[i];
}

// ---------------------------------------------------------------------------
// Layer-0 input projection: xg[m, g] = b[g] + sum_d embeds[b, t, d] * Wih[g, d]
//   m = t*cB + b  (folds the (B,T,D)->(T,B,D) transpose)
// grid: (cM/64, cG/64), block 128 (4 waves); each wave = 16 rows x 64 cols
// ---------------------------------------------------------------------------
__global__ void __launch_bounds__(128)
k_proj_l0(const float* __restrict__ X, const _Float16* __restrict__ W,
          const float* __restrict__ bias, float* __restrict__ out) {
  const int lane = threadIdx.x & 31;
  const int wave = threadIdx.x >> 5;
  const int half = lane >> 4;
  const int l15  = lane & 15;
  const int m0 = blockIdx.x * 64 + wave * 16;
  const int n0 = blockIdx.y * 64;

  const int row = m0 + l15;            // A-frag row (= t*cB + b)
  const int t = row >> 6;              // cB == 64
  const int b = row & 63;
  const float* xrow = X + ((size_t)b * cT + t) * cD;

  v8f acc[4];
#pragma unroll
  for (int nt = 0; nt < 4; ++nt)
#pragma unroll
    for (int r = 0; r < 8; ++r) acc[nt][r] = 0.0f;

  for (int k = 0; k < cD; k += 32) {
    v16h a = frag_a_f(xrow, k, half);
#pragma unroll
    for (int nt = 0; nt < 4; ++nt) {
      const _Float16* wcol = W + (size_t)(n0 + nt * 16 + l15) * cD;
      acc[nt] = wmma_f16(a, frag_b_h(wcol, k, half), acc[nt]);
    }
  }

#pragma unroll
  for (int nt = 0; nt < 4; ++nt) {
    const int col = n0 + nt * 16 + l15;
    const float bv = bias[col];
#pragma unroll
    for (int r = 0; r < 8; ++r)
      out[(size_t)(m0 + r + 8 * half) * cG + col] = acc[nt][r] + bv;
  }
}

// ---------------------------------------------------------------------------
// Layer-1 input projection: xg[m, g] = b[g] + sum_j h0[m, j] * Wih[g, j], K=256
// ---------------------------------------------------------------------------
__global__ void __launch_bounds__(128)
k_proj_l1(const _Float16* __restrict__ X, const _Float16* __restrict__ W,
          const float* __restrict__ bias, float* __restrict__ out) {
  const int lane = threadIdx.x & 31;
  const int wave = threadIdx.x >> 5;
  const int half = lane >> 4;
  const int l15  = lane & 15;
  const int m0 = blockIdx.x * 64 + wave * 16;
  const int n0 = blockIdx.y * 64;

  const _Float16* xrow = X + (size_t)(m0 + l15) * cK2;

  v8f acc[4];
#pragma unroll
  for (int nt = 0; nt < 4; ++nt)
#pragma unroll
    for (int r = 0; r < 8; ++r) acc[nt][r] = 0.0f;

#pragma unroll
  for (int k = 0; k < cK2; k += 32) {
    v16h a = frag_a_h(xrow, k, half);
#pragma unroll
    for (int nt = 0; nt < 4; ++nt) {
      const _Float16* wcol = W + (size_t)(n0 + nt * 16 + l15) * cK2;
      acc[nt] = wmma_f16(a, frag_b_h(wcol, k, half), acc[nt]);
    }
  }

#pragma unroll
  for (int nt = 0; nt < 4; ++nt) {
    const int col = n0 + nt * 16 + l15;
    const float bv = bias[col];
#pragma unroll
    for (int r = 0; r < 8; ++r)
      out[(size_t)(m0 + r + 8 * half) * cG + col] = acc[nt][r] + bv;
  }
}

// ---------------------------------------------------------------------------
// Persistent bidirectional LSTM layer. grid = 2 blocks (0=fwd, 1=bwd),
// block = 512 threads (16 waves). Each wave owns 2 tasks; a task = one
// 16-row batch tile x one 16-wide h-group, covering all four gate tiles
// (cols hg*16+l15 + {0,128,256,384}) so the (c,h) update stays in registers.
// h state lives in LDS as f16 (A-fragment source); c state in registers.
// ---------------------------------------------------------------------------
__global__ void __launch_bounds__(512)
k_lstm_layer(const float* __restrict__ xgF, const float* __restrict__ xgB,
             const _Float16* __restrict__ whhF, const _Float16* __restrict__ whhB,
             _Float16* __restrict__ hOut) {
  __shared__ _Float16 hsh[cB * cH];   // 16 KB

  const int dir = blockIdx.x;
  const float*    xg  = dir ? xgB : xgF;
  const _Float16* whh = dir ? whhB : whhF;

  for (int i = threadIdx.x; i < cB * cH; i += 512) hsh[i] = (_Float16)0.0f;

  const int lane = threadIdx.x & 31;
  const int wave = threadIdx.x >> 5;
  const int half = lane >> 4;
  const int l15  = lane & 15;

  float creg[2][8];
#pragma unroll
  for (int q = 0; q < 2; ++q)
#pragma unroll
    for (int r = 0; r < 8; ++r) creg[q][r] = 0.0f;

  __syncthreads();

  for (int s = 0; s < cT; ++s) {
    const int t = dir ? (cT - 1 - s) : s;
    const float* xgt = xg + (size_t)t * (cB * cG);

    v8f acc[2][4];
    // ---- phase 1: gates = xg[t] + h @ Whh^T  (reads hsh) ----
#pragma unroll
    for (int q = 0; q < 2; ++q) {
      const int tt = wave * 2 + q;       // 0..31
      const int mt = tt >> 3;            // batch tile 0..3
      const int hg = tt & 7;             // h group 0..7
      const _Float16* arow = hsh + (size_t)(mt * 16 + l15) * cH;
#pragma unroll
      for (int g = 0; g < 4; ++g) {
        const int col = g * cH + hg * 16 + l15;
#pragma unroll
        for (int r = 0; r < 8; ++r)
          acc[q][g][r] = xgt[(size_t)(mt * 16 + r + 8 * half) * cG + col];
      }
#pragma unroll
      for (int k = 0; k < cH; k += 32) {
        v16h a = frag_a_h(arow, k, half);
#pragma unroll
        for (int g = 0; g < 4; ++g) {
          const int col = g * cH + hg * 16 + l15;
          const _Float16* wcol = whh + (size_t)col * cH;
          acc[q][g] = wmma_f16(a, frag_b_h(wcol, k, half), acc[q][g]);
        }
      }
    }
    __syncthreads();

    // ---- phase 2: c,h update in registers; write new h to LDS + global ----
#pragma unroll
    for (int q = 0; q < 2; ++q) {
      const int tt = wave * 2 + q;
      const int mt = tt >> 3;
      const int hg = tt & 7;
      const int h = hg * 16 + l15;
#pragma unroll
      for (int r = 0; r < 8; ++r) {
        const int b = mt * 16 + r + 8 * half;
        const float gi = acc[q][0][r];
        const float gf = acc[q][1][r];
        const float gg = acc[q][2][r];
        const float go = acc[q][3][r];
        const float c  = sigf(gf) * creg[q][r] + sigf(gi) * tanhf(gg);
        const float hn = sigf(go) * tanhf(c);
        creg[q][r] = c;
        const _Float16 hf = (_Float16)hn;
        hsh[b * cH + h] = hf;
        hOut[((size_t)t * cB + b) * cK2 + dir * cH + h] = hf;
      }
    }
    __syncthreads();
  }
}

// ---------------------------------------------------------------------------
// Emission: emis[m, k] = blin[k] + sum_j h1[m, j] * wlin[k, j]   (K=11, tiny)
// ---------------------------------------------------------------------------
__global__ void k_emis(const _Float16* __restrict__ h1,
                       const float* __restrict__ wlin,
                       const float* __restrict__ blin,
                       float* __restrict__ emis) {
  int gid = blockIdx.x * blockDim.x + threadIdx.x;
  if (gid >= cM * cNK) return;
  int m = gid / cNK;
  int k = gid - m * cNK;
  const _Float16* hp = h1 + (size_t)m * cK2;
  const float*    wp = wlin + (size_t)k * cK2;
  float s = blin[k];
  for (int j = 0; j < cK2; ++j) s += (float)hp[j] * wp[j];
  emis[gid] = s;
}

// ---------------------------------------------------------------------------
// CRF NLL (mask is all-ones in this problem). One thread per batch row.
// emis layout: [t][b][k].  tags layout: [b][t].
// ---------------------------------------------------------------------------
__global__ void __launch_bounds__(64)
k_crf(const float* __restrict__ emis, const int* __restrict__ tags,
      const float* __restrict__ trans, const float* __restrict__ startT,
      const float* __restrict__ endT, float* __restrict__ out) {
  __shared__ float tr[cNK * cNK];
  __shared__ float st[cNK];
  __shared__ float en[cNK];
  __shared__ float red[cB];

  for (int i = threadIdx.x; i < cNK * cNK; i += 64) tr[i] = trans[i];
  for (int i = threadIdx.x; i < cNK; i += 64) { st[i] = startT[i]; en[i] = endT[i]; }
  __syncthreads();

  const int b = threadIdx.x;           // 0..63
  const int* tg = tags + (size_t)b * cT;

  float alpha[cNK], nxt[cNK], ev[cNK];
#pragma unroll
  for (int k = 0; k < cNK; ++k) {
    ev[k] = emis[(size_t)b * cNK + k];           // t = 0
    alpha[k] = st[k] + ev[k];
  }
  int prev = tg[0];
  float gold = st[prev] + ev[prev];

  for (int t = 1; t < cT; ++t) {
    const float* e = emis + ((size_t)t * cB + b) * cNK;
#pragma unroll
    for (int k = 0; k < cNK; ++k) ev[k] = e[k];
    const int tag = tg[t];
    gold += ev[tag] + tr[prev * cNK + tag];
    prev = tag;
#pragma unroll
    for (int j = 0; j < cNK; ++j) {
      float mx = -3.0e38f;
#pragma unroll
      for (int i = 0; i < cNK; ++i) mx = fmaxf(mx, alpha[i] + tr[i * cNK + j]);
      float s = 0.0f;
#pragma unroll
      for (int i = 0; i < cNK; ++i) s += __expf(alpha[i] + tr[i * cNK + j] - mx);
      nxt[j] = mx + __logf(s) + ev[j];
    }
#pragma unroll
    for (int j = 0; j < cNK; ++j) alpha[j] = nxt[j];
  }
  gold += en[prev];

  float mx = -3.0e38f;
#pragma unroll
  for (int k = 0; k < cNK; ++k) mx = fmaxf(mx, alpha[k] + en[k]);
  float s = 0.0f;
#pragma unroll
  for (int k = 0; k < cNK; ++k) s += __expf(alpha[k] + en[k] - mx);
  const float logZ = mx + __logf(s);

  red[b] = gold - logZ;
  __syncthreads();
  if (b == 0) {
    float acc = 0.0f;
    for (int i = 0; i < cB; ++i) acc += red[i];
    out[0] = -acc / (float)cB;
  }
}

// ---------------------------------------------------------------------------
// Host driver
// ---------------------------------------------------------------------------
extern "C" void kernel_launch(void* const* d_in, const int* in_sizes, int n_in,
                              void* d_out, int out_size, void* d_ws,
                              size_t ws_size, hipStream_t stream) {
  (void)in_sizes; (void)n_in; (void)out_size; (void)ws_size;

  const float* embeds = (const float*)d_in[0];
  const int*   tags   = (const int*)d_in[1];
  // d_in[2] = mask (all ones in this problem; unused)
  const float* wih00 = (const float*)d_in[3];
  const float* whh00 = (const float*)d_in[4];
  const float* b00   = (const float*)d_in[5];
  const float* wih01 = (const float*)d_in[6];
  const float* whh01 = (const float*)d_in[7];
  const float* b01   = (const float*)d_in[8];
  const float* wih10 = (const float*)d_in[9];
  const float* whh10 = (const float*)d_in[10];
  const float* b10   = (const float*)d_in[11];
  const float* wih11 = (const float*)d_in[12];
  const float* whh11 = (const float*)d_in[13];
  const float* b11   = (const float*)d_in[14];
  const float* wlin  = (const float*)d_in[15];
  const float* blin  = (const float*)d_in[16];
  const float* trans = (const float*)d_in[17];
  const float* startT= (const float*)d_in[18];
  const float* endT  = (const float*)d_in[19];

  char* ws = (char*)d_ws;
  size_t off = 0;
  auto alloc = [&](size_t bytes) -> char* {
    char* p = ws + off;
    off = (off + bytes + 255) & ~(size_t)255;
    return p;
  };

  _Float16* wih0fH = (_Float16*)alloc((size_t)cG * cD * 2);
  _Float16* wih0bH = (_Float16*)alloc((size_t)cG * cD * 2);
  _Float16* whh0fH = (_Float16*)alloc((size_t)cG * cH * 2);
  _Float16* whh0bH = (_Float16*)alloc((size_t)cG * cH * 2);
  _Float16* wih1fH = (_Float16*)alloc((size_t)cG * cK2 * 2);
  _Float16* wih1bH = (_Float16*)alloc((size_t)cG * cK2 * 2);
  _Float16* whh1fH = (_Float16*)alloc((size_t)cG * cH * 2);
  _Float16* whh1bH = (_Float16*)alloc((size_t)cG * cH * 2);
  float*    xgf    = (float*)alloc((size_t)cM * cG * 4);
  float*    xgb    = (float*)alloc((size_t)cM * cG * 4);
  _Float16* h0h    = (_Float16*)alloc((size_t)cM * cK2 * 2);
  _Float16* h1h    = (_Float16*)alloc((size_t)cM * cK2 * 2);
  float*    emis   = (float*)alloc((size_t)cM * cNK * 4);

  auto conv = [&](const float* s, _Float16* d, int n) {
    k_f32_to_f16<<<(n + 255) / 256, 256, 0, stream>>>(s, d, n);
  };
  conv(wih00, wih0fH, cG * cD);
  conv(wih01, wih0bH, cG * cD);
  conv(whh00, whh0fH, cG * cH);
  conv(whh01, whh0bH, cG * cH);
  conv(wih10, wih1fH, cG * cK2);
  conv(wih11, wih1bH, cG * cK2);
  conv(whh10, whh1fH, cG * cH);
  conv(whh11, whh1bH, cG * cH);

  dim3 pg(cM / 64, cG / 64);   // (512, 8)
  k_proj_l0<<<pg, 128, 0, stream>>>(embeds, wih0fH, b00, xgf);
  k_proj_l0<<<pg, 128, 0, stream>>>(embeds, wih0bH, b01, xgb);

  k_lstm_layer<<<2, 512, 0, stream>>>(xgf, xgb, whh0fH, whh0bH, h0h);

  k_proj_l1<<<pg, 128, 0, stream>>>(h0h, wih1fH, b10, xgf);
  k_proj_l1<<<pg, 128, 0, stream>>>(h0h, wih1bH, b11, xgb);

  k_lstm_layer<<<2, 512, 0, stream>>>(xgf, xgb, whh1fH, whh1bH, h1h);

  k_emis<<<(cM * cNK + 255) / 256, 256, 0, stream>>>(h1h, wlin, blin, emis);

  k_crf<<<1, 64, 0, stream>>>(emis, tags, trans, startT, endT, (float*)d_out);
}
